// GroupedQueryAttention_10874857193517
// MI455X (gfx1250) — matrix operand order repaired and checked by reading, not compiled, not run on hardware
//
#include <hip/hip_runtime.h>
#include <hip/hip_bf16.h>
#include <math.h>

typedef __bf16 bf16;
typedef __attribute__((ext_vector_type(16))) __bf16 v16bf;
typedef __attribute__((ext_vector_type(8)))  __bf16 v8bf;
typedef __attribute__((ext_vector_type(8)))  float  v8f;
typedef __attribute__((ext_vector_type(4)))  unsigned v4u;
typedef __attribute__((ext_vector_type(8)))  int      v8i;
typedef __attribute__((ext_vector_type(4)))  int      v4i;

static __device__ __forceinline__ bf16 f2bf(float f) {
  union { float f; unsigned u; } a; a.f = f;
  unsigned r = a.u + 0x7FFFu + ((a.u >> 16) & 1u);   // round-to-nearest-even
  union { unsigned short s; bf16 b; } o; o.s = (unsigned short)(r >> 16);
  return o.b;
}

static __device__ __forceinline__ v16bf cat8(v8bf lo, v8bf hi) {
  return __builtin_shufflevector(lo, hi, 0,1,2,3,4,5,6,7,8,9,10,11,12,13,14,15);
}

#if __has_builtin(__builtin_amdgcn_tensor_load_to_lds)
#define HAVE_TDM 1
#else
#define HAVE_TDM 0
#endif

#if HAVE_TDM
// 2D tile load via Tensor Data Mover: tile_d1 rows x tile_d0 elems (bf16),
// source rows strided by width_elems. D# layout per CDNA5 ISA ch.8.
// Toolchain: 6-arg builtin (g0 v4u, g1 v8i, g2 v4i, g3 v4i, v8i, cpol).
static __device__ __forceinline__ void tdm_load_2d(
    unsigned lds_addr, const void* gaddr, unsigned tile_d0, unsigned tile_d1,
    unsigned width_elems, unsigned height_rows) {
  unsigned long long ga = (unsigned long long)(size_t)gaddr;
  v4u g0;
  g0[0] = 1u;                                        // count=1, user mode
  g0[1] = lds_addr;                                  // lds_addr[31:0]
  g0[2] = (unsigned)ga;                              // global_addr[31:0]
  g0[3] = (unsigned)((ga >> 32) & 0x01FFFFFFull)     // global_addr[56:32]
          | (2u << 30);                              // type = 2 (image)
  v8i g1;
  g1[0] = (int)(1u << 16);                           // data_size=1 -> 2 bytes
  g1[1] = (int)((width_elems & 0xFFFFu) << 16);      // tensor_dim0[15:0]
  g1[2] = (int)(((width_elems >> 16) & 0xFFFFu)      // tensor_dim0[31:16]
                | ((height_rows & 0xFFFFu) << 16));  // tensor_dim1[15:0]
  g1[3] = (int)(((height_rows >> 16) & 0xFFFFu)      // tensor_dim1[31:16]
                | (tile_d0 << 16));                  // tile_dim0
  g1[4] = (int)tile_d1;                              // tile_dim1 (tile_dim2=0)
  g1[5] = (int)width_elems;                          // tensor_dim0_stride[31:0]
  g1[6] = 0;                                         // stride hi / dim1_stride
  g1[7] = 0;
  v4i z4 = (v4i){0, 0, 0, 0};
  v8i z8 = (v8i){0, 0, 0, 0, 0, 0, 0, 0};
  __builtin_amdgcn_tensor_load_to_lds(g0, g1, z4, z4, z8, 0);
}
#endif

// ---------------------------------------------------------------- converts
__global__ void cvt_f32_bf16(const float* __restrict__ in, bf16* __restrict__ out,
                             long long n) {
  long long i = (long long)blockIdx.x * blockDim.x + threadIdx.x;
  if (i < n) out[i] = f2bf(in[i]);
}

// out[n*K + k] = bf16(in[k*N + n])  — tiled transpose (done once per weight)
__global__ __launch_bounds__(256) void cvt_transpose_bf16(
    const float* __restrict__ in, bf16* __restrict__ out, int K, int N) {
  __shared__ float t[32][33];
  const int k0 = blockIdx.y * 32, n0 = blockIdx.x * 32;
  const int tx = threadIdx.x, ty = threadIdx.y;   // 32 x 8
#pragma unroll
  for (int i = 0; i < 32; i += 8)
    t[ty + i][tx] = in[(size_t)(k0 + ty + i) * N + n0 + tx];
  __syncthreads();
#pragma unroll
  for (int i = 0; i < 32; i += 8)
    out[(size_t)(n0 + ty + i) * K + k0 + tx] = f2bf(t[tx][ty + i]);
}

// ---------------------------------------------------------------- GEMM
// C[M,N] = A[M,K] * Bt[N,K]^T + bias[N].  A,Bt bf16 row-major; C f32 or bf16.
// Block tile 128x256, 8 waves (2 in M x 4 in N), wave tile 64x64, K step 32.
template <int OUT_BF16>
__global__ __launch_bounds__(256) void gemm_wmma_bf16(
    const bf16* __restrict__ A, const bf16* __restrict__ Bt,
    const float* __restrict__ bias, void* __restrict__ Cout,
    int M, int N, int K) {
  __shared__ bf16 As[128 * 32];   // [m][k]
  __shared__ bf16 Bs[256 * 32];   // [n][k]
  const int tid  = threadIdx.x;
  const int lane = tid & 31, wid = tid >> 5;
  const int half = lane >> 4, l16 = lane & 15;
  const int wm = wid >> 2, wn = wid & 3;
  const int m0 = blockIdx.y * 128, n0 = blockIdx.x * 256;

  v8f acc[4][4];
#pragma unroll
  for (int mi = 0; mi < 4; ++mi)
#pragma unroll
    for (int ni = 0; ni < 4; ++ni)
      acc[mi][ni] = (v8f){0.f,0.f,0.f,0.f,0.f,0.f,0.f,0.f};

#if HAVE_TDM
  const unsigned as_off = (unsigned)(size_t)(void*)As;
  const unsigned bs_off = (unsigned)(size_t)(void*)Bs;
#endif

  for (int k0 = 0; k0 < K; k0 += 32) {
    __syncthreads();   // previous tiles fully consumed
#if HAVE_TDM
    if (wid == 0) {
      tdm_load_2d(as_off, A  + (size_t)m0 * K + k0, 32u, 128u,
                  (unsigned)K, (unsigned)M);
      tdm_load_2d(bs_off, Bt + (size_t)n0 * K + k0, 32u, 256u,
                  (unsigned)K, (unsigned)N);
      __builtin_amdgcn_s_wait_tensorcnt(0);
    }
#else
#pragma unroll
    for (int it = 0; it < 2; ++it) {            // A: 512 16B chunks
      int c = tid + it * 256;
      int r = c >> 2, cc = c & 3;
      *(uint4*)(&As[r * 32 + cc * 8]) =
          *(const uint4*)(&A[(size_t)(m0 + r) * K + k0 + cc * 8]);
    }
#pragma unroll
    for (int it = 0; it < 4; ++it) {            // B: 1024 16B chunks
      int c = tid + it * 256;
      int r = c >> 2, cc = c & 3;
      *(uint4*)(&Bs[r * 32 + cc * 8]) =
          *(const uint4*)(&Bt[(size_t)(n0 + r) * K + k0 + cc * 8]);
    }
#endif
    __syncthreads();

    v16bf af[4];
#pragma unroll
    for (int mi = 0; mi < 4; ++mi) {
      int mr = wm * 64 + mi * 16 + l16;
      v8bf lo = *(const v8bf*)(&As[mr * 32 + half * 8]);        // A: k=half*8..
      v8bf hi = *(const v8bf*)(&As[mr * 32 + 16 + half * 8]);   //    k=16+half*8..
      af[mi] = cat8(lo, hi);
    }
#pragma unroll
    for (int ni = 0; ni < 4; ++ni) {
      int nr = wn * 64 + ni * 16 + l16;
      v8bf lo = *(const v8bf*)(&Bs[nr * 32 + half * 8]);        // B: k=half*8..
      v8bf hi = *(const v8bf*)(&Bs[nr * 32 + 16 + half * 8]);
      v16bf bfr = cat8(lo, hi);
#pragma unroll
      for (int mi = 0; mi < 4; ++mi)
        acc[mi][ni] = __builtin_amdgcn_wmma_f32_16x16x32_bf16(
            false, af[mi], false, bfr, (short)0, acc[mi][ni], false, false);
    }
  }

#pragma unroll
  for (int ni = 0; ni < 4; ++ni) {
    int nn = n0 + wn * 64 + ni * 16 + l16;
    float bv = bias ? bias[nn] : 0.0f;
#pragma unroll
    for (int mi = 0; mi < 4; ++mi) {
#pragma unroll
      for (int v = 0; v < 8; ++v) {
        int mm = m0 + wm * 64 + mi * 16 + v + 8 * half;  // C layout: m=v+8*half
        float val = acc[mi][ni][v] + bv;
        if (OUT_BF16)
          ((bf16*)Cout)[(size_t)mm * N + nn] = f2bf(val);
        else
          ((float*)Cout)[(size_t)mm * N + nn] = val;
      }
    }
  }
}

// ---------------------------------------------------------------- attention
// Q: [B,S,2048] bf16 (head h at col h*128). KV: [B,S,1024] (K:0..511, V:512..1023).
// O: [B,S,2048] bf16. Grid: (S/128, B*16). Block: 8 waves; wave = 16 query rows.
__global__ __launch_bounds__(256) void gqa_flash_attn(
    const bf16* __restrict__ Q, const bf16* __restrict__ KV,
    bf16* __restrict__ O) {
  constexpr int S = 2048, D = 2048, KVD = 1024, HD = 128;
  __shared__ bf16 Ks[32 * 128];     // [key][d]
  __shared__ bf16 Vs[128 * 32];     // [d][key]  (transposed)
  __shared__ bf16 Ps[8][16 * 32];   // per-wave P tile [m][key]
  const int tid = threadIdx.x, lane = tid & 31, wid = tid >> 5;
  const int half = lane >> 4, l16 = lane & 15;
  const int bh = blockIdx.y, b = bh >> 4, h = bh & 15, hkv = h >> 2;
  const int qb = blockIdx.x;
  const int qrow0 = qb * 128 + wid * 16;

  const bf16* qp = Q + ((size_t)(b * S + qrow0 + l16)) * D + h * HD;
  v16bf qa[4];
#pragma unroll
  for (int c = 0; c < 4; ++c) {
    v8bf lo = *(const v8bf*)(qp + c * 32 + half * 8);
    v8bf hi = *(const v8bf*)(qp + c * 32 + 16 + half * 8);
    qa[c] = cat8(lo, hi);
  }

  v8f o[8];
  float mrow[8], lrow[8];
#pragma unroll
  for (int ct = 0; ct < 8; ++ct) o[ct] = (v8f){0.f,0.f,0.f,0.f,0.f,0.f,0.f,0.f};
#pragma unroll
  for (int v = 0; v < 8; ++v) { mrow[v] = -1e30f; lrow[v] = 0.0f; }

  const bf16* kb = KV + (size_t)b * S * KVD + hkv * HD;
  const bf16* vb = kb + 512;
  const int nkt = qb * 4 + 4;                 // causal: keys up to (qb+1)*128
  const float scale = 0.08838834764831845f;   // 1/sqrt(128)

  for (int kt = 0; kt < nkt; ++kt) {
    const int key0 = kt * 32;
#pragma unroll
    for (int it = 0; it < 2; ++it) {          // K tile 32x128, b128 chunks
      int c = tid + it * 256;
      int r = c >> 4, cc = c & 15;
      *(uint4*)(&Ks[r * 128 + cc * 8]) =
          *(const uint4*)(kb + (size_t)(key0 + r) * KVD + cc * 8);
    }
#pragma unroll
    for (int i = 0; i < 16; ++i) {            // V tile transposed -> Vs[d][key]
      int e = tid + i * 256;
      int kk = e >> 7, dd = e & 127;
      Vs[dd * 32 + kk] = vb[(size_t)(key0 + kk) * KVD + dd];
    }
    __syncthreads();

    v8f s0 = (v8f){0.f,0.f,0.f,0.f,0.f,0.f,0.f,0.f};
    v8f s1 = (v8f){0.f,0.f,0.f,0.f,0.f,0.f,0.f,0.f};
#pragma unroll
    for (int c = 0; c < 4; ++c) {
      v8bf lo0 = *(const v8bf*)(&Ks[l16 * 128 + c * 32 + half * 8]);
      v8bf hi0 = *(const v8bf*)(&Ks[l16 * 128 + c * 32 + 16 + half * 8]);
      v8bf lo1 = *(const v8bf*)(&Ks[(16 + l16) * 128 + c * 32 + half * 8]);
      v8bf hi1 = *(const v8bf*)(&Ks[(16 + l16) * 128 + c * 32 + 16 + half * 8]);
      s0 = __builtin_amdgcn_wmma_f32_16x16x32_bf16(
          false, qa[c], false, cat8(lo0, hi0), (short)0, s0, false, false);
      s1 = __builtin_amdgcn_wmma_f32_16x16x32_bf16(
          false, qa[c], false, cat8(lo1, hi1), (short)0, s1, false, false);
    }

    const int kg0 = key0 + l16, kg1 = key0 + 16 + l16;
#pragma unroll
    for (int v = 0; v < 8; ++v) {
      const int qg = qrow0 + v + 8 * half;
      float x0 = (kg0 <= qg) ? s0[v] * scale : -1e30f;
      float x1 = (kg1 <= qg) ? s1[v] * scale : -1e30f;
      float rm = fmaxf(x0, x1);
      rm = fmaxf(rm, __shfl_xor(rm, 1));
      rm = fmaxf(rm, __shfl_xor(rm, 2));
      rm = fmaxf(rm, __shfl_xor(rm, 4));
      rm = fmaxf(rm, __shfl_xor(rm, 8));
      float mn = fmaxf(mrow[v], rm);
      float alpha = __expf(mrow[v] - mn);
      mrow[v] = mn;
      float p0 = __expf(x0 - mn);
      float p1 = __expf(x1 - mn);
      float rs = p0 + p1;
      rs += __shfl_xor(rs, 1);
      rs += __shfl_xor(rs, 2);
      rs += __shfl_xor(rs, 4);
      rs += __shfl_xor(rs, 8);
      lrow[v] = lrow[v] * alpha + rs;
#pragma unroll
      for (int ct = 0; ct < 8; ++ct) o[ct][v] *= alpha;
      const int m = v + 8 * half;
      Ps[wid][m * 32 + l16]      = f2bf(p0);
      Ps[wid][m * 32 + 16 + l16] = f2bf(p1);
    }
    asm volatile("s_wait_dscnt 0x0" ::: "memory");  // wave-private LDS RAW

    v8bf plo = *(const v8bf*)(&Ps[wid][l16 * 32 + half * 8]);
    v8bf phi = *(const v8bf*)(&Ps[wid][l16 * 32 + 16 + half * 8]);
    v16bf pa = cat8(plo, phi);
#pragma unroll
    for (int ct = 0; ct < 8; ++ct) {
      v8bf lo = *(const v8bf*)(&Vs[(ct * 16 + l16) * 32 + half * 8]);
      v8bf hi = *(const v8bf*)(&Vs[(ct * 16 + l16) * 32 + 16 + half * 8]);
      o[ct] = __builtin_amdgcn_wmma_f32_16x16x32_bf16(
          false, pa, false, cat8(lo, hi), (short)0, o[ct], false, false);
    }
    __syncthreads();
  }

  bf16* op = O + (size_t)(b * S) * D + h * HD;
#pragma unroll
  for (int ct = 0; ct < 8; ++ct) {
    int nn = ct * 16 + l16;
#pragma unroll
    for (int v = 0; v < 8; ++v) {
      int mg = qrow0 + v + 8 * half;
      op[(size_t)mg * D + nn] = f2bf(o[ct][v] / lrow[v]);
    }
  }
}

// ---------------------------------------------------------------- launch
extern "C" void kernel_launch(void* const* d_in, const int* in_sizes, int n_in,
                              void* d_out, int out_size, void* d_ws, size_t ws_size,
                              hipStream_t stream) {
  const float* x   = (const float*)d_in[0];
  const float* Wq  = (const float*)d_in[1];
  const float* bq  = (const float*)d_in[2];
  const float* Wkv = (const float*)d_in[3];
  const float* bkv = (const float*)d_in[4];
  const float* Wc  = (const float*)d_in[5];
  const float* bc  = (const float*)d_in[6];

  const int B = 4, S = 2048, D = 2048, KVD = 1024;
  const long long M = (long long)B * S;   // 8192

  char* ws = (char*)d_ws;
  size_t off = 0;
  bf16* xb   = (bf16*)(ws + off); off += (size_t)M * D * 2;       // x bf16
  bf16* Wqt  = (bf16*)(ws + off); off += (size_t)D * D * 2;       // Wq^T bf16 [N,K]
  bf16* Wkvt = (bf16*)(ws + off); off += (size_t)D * KVD * 2;     // Wkv^T bf16
  bf16* Wct  = (bf16*)(ws + off); off += (size_t)D * D * 2;       // Wc^T bf16
  bf16* Qb   = (bf16*)(ws + off); off += (size_t)M * D * 2;
  bf16* KVb  = (bf16*)(ws + off); off += (size_t)M * KVD * 2;
  bf16* Ob   = (bf16*)(ws + off); off += (size_t)M * D * 2;
  (void)ws_size; (void)in_sizes; (void)n_in; (void)out_size;

  long long nx = M * D;
  cvt_f32_bf16<<<(unsigned)((nx + 255) / 256), 256, 0, stream>>>(x, xb, nx);
  dim3 tb(32, 8);
  cvt_transpose_bf16<<<dim3(D / 32, D / 32), tb, 0, stream>>>(Wq, Wqt, D, D);
  cvt_transpose_bf16<<<dim3(KVD / 32, D / 32), tb, 0, stream>>>(Wkv, Wkvt, D, KVD);
  cvt_transpose_bf16<<<dim3(D / 32, D / 32), tb, 0, stream>>>(Wc, Wct, D, D);

  dim3 blk(256);
  // Q = x*Wq + bq   (bf16 out, [B,S,D])
  gemm_wmma_bf16<1><<<dim3(D / 256, (int)(M / 128)), blk, 0, stream>>>(
      xb, Wqt, bq, Qb, (int)M, D, D);
  // KV = x*Wkv + bkv (bf16 out, [B,S,1024])
  gemm_wmma_bf16<1><<<dim3(KVD / 256, (int)(M / 128)), blk, 0, stream>>>(
      xb, Wkvt, bkv, KVb, (int)M, KVD, D);
  // flash attention -> Ob [B,S,D]
  gqa_flash_attn<<<dim3(S / 128, B * 16), blk, 0, stream>>>(Qb, KVb, Ob);
  // out = Ob*Wc + bc (f32 out)
  gemm_wmma_bf16<0><<<dim3(D / 256, (int)(M / 128)), blk, 0, stream>>>(
      Ob, Wct, bc, d_out, (int)M, D, D);
}
